// Block_rel_8804682956947
// MI455X (gfx1250) — compile-verified, matrix operand order checked by reading, not run on hardware
//
#include <hip/hip_runtime.h>
#include <hip/hip_bf16.h>
#include <math.h>

typedef __bf16 bf16;
typedef bf16  bf16x16 __attribute__((ext_vector_type(16)));
typedef bf16  bf16x8  __attribute__((ext_vector_type(8)));
typedef float f32x8   __attribute__((ext_vector_type(8)));

union ABFrag { bf16x16 v; bf16 e[16]; bf16x8 h[2]; };

__device__ __forceinline__ f32x8 wmma_bf16(const ABFrag& a, const ABFrag& b, f32x8 c) {
    return __builtin_amdgcn_wmma_f32_16x16x32_bf16(false, a.v, false, b.v, (short)0, c, false, false);
}

// Async global->LDS copy of 16 bytes (gfx1250 GLOBAL_LOAD_ASYNC_TO_LDS_B128,
// tracked by ASYNCcnt). LDS generic pointers hold the LDS byte offset in the
// low 32 bits (flat-aperture rule, ISA 10.2).
__device__ __forceinline__ void async_copy_b128(void* lds_dst, const void* gsrc) {
    unsigned lds_addr = (unsigned)(unsigned long long)lds_dst;
    asm volatile("global_load_async_to_lds_b128 %0, %1, off"
                 :: "v"(lds_addr), "v"(gsrc) : "memory");
}
__device__ __forceinline__ void async_wait0() {
    asm volatile("s_wait_asynccnt 0x0" ::: "memory");
}

// ---------------------------------------------------------------------------
// Elementwise fp32 -> bf16 conversion (weights, etc.)
// ---------------------------------------------------------------------------
__global__ void cvt_bf16_kernel(const float* __restrict__ src, bf16* __restrict__ dst, int n) {
    for (int i = blockIdx.x * blockDim.x + threadIdx.x; i < n; i += gridDim.x * blockDim.x)
        dst[i] = (bf16)src[i];
}

// ---------------------------------------------------------------------------
// LayerNorm over last dim (D=384), output bf16. One row per 128-thread block.
// ---------------------------------------------------------------------------
__global__ __launch_bounds__(128)
void ln_kernel(const float* __restrict__ x, const float* __restrict__ w,
               const float* __restrict__ bv, bf16* __restrict__ out) {
    __shared__ float rs[128], rq[128];
    const int row = blockIdx.x, tid = threadIdx.x;
    const float* xr = x + (long long)row * 384;
    float v[3], s = 0.f, q = 0.f;
#pragma unroll
    for (int t = 0; t < 3; ++t) { v[t] = xr[tid + t * 128]; s += v[t]; q += v[t] * v[t]; }
    rs[tid] = s; rq[tid] = q; __syncthreads();
    for (int o = 64; o > 0; o >>= 1) {
        if (tid < o) { rs[tid] += rs[tid + o]; rq[tid] += rq[tid + o]; }
        __syncthreads();
    }
    const float mean = rs[0] * (1.f / 384.f);
    const float var  = rq[0] * (1.f / 384.f) - mean * mean;
    const float inv  = rsqrtf(var + 1e-5f);
    bf16* orow = out + (long long)row * 384;
#pragma unroll
    for (int t = 0; t < 3; ++t) {
        const int j = tid + t * 128;
        orow[j] = (bf16)((v[t] - mean) * inv * w[j] + bv[j]);
    }
}

// ---------------------------------------------------------------------------
// Transpose v: (B,N,H*64) bf16 -> vT (B,H,64,N) bf16
// ---------------------------------------------------------------------------
__global__ void transpose_v_kernel(const bf16* __restrict__ v, bf16* __restrict__ vT, int n) {
    for (int idx = blockIdx.x * blockDim.x + threadIdx.x; idx < n; idx += gridDim.x * blockDim.x) {
        const int j = idx & 511;
        const int c = (idx >> 9) & 63;
        const int h = (idx >> 15) % 6;
        const int b = idx / (512 * 64 * 6);
        vT[idx] = v[((long long)(b * 512 + j)) * 384 + h * 64 + c];
    }
}

// ---------------------------------------------------------------------------
// Generic batched WMMA GEMM:  C = alpha*(A @ Bt^T) [+bias] [gelu] [+Add]
//   A  : M x K bf16, row stride ldA
//   Bt : N x K bf16 (i.e., B transposed), row stride ldB
//   C  : M x N fp32 or bf16, row stride ldC
// Per-batch (blockIdx.z): zo = z/innerCount, zi = z%innerCount; each operand
// base advances by zo*Out + zi*In elements. 64x64 tile / workgroup, 8 waves,
// each wave owns two 16x16 subtiles; K staged through LDS in steps of 32 via
// async global->LDS b128 copies. Requires M%64==0, N%64==0, K%32==0.
// ---------------------------------------------------------------------------
template<bool BIAS, bool GELU, bool BF16OUT, bool ADDIN>
__global__ __launch_bounds__(256)
void gemm_bf16_kernel(const bf16* __restrict__ A,  int ldA, long long aOut, long long aIn,
                      const bf16* __restrict__ Bt, int ldB, long long bOut, long long bIn,
                      void* __restrict__ Cp,       int ldC, long long cOut, long long cIn,
                      const float* __restrict__ bias,
                      const float* __restrict__ Add, int ldAdd, long long dOut, long long dIn,
                      int innerCount, int K, float alpha) {
    __shared__ bf16 As[64 * 32];
    __shared__ bf16 Bs[64 * 32];
    const int tid = threadIdx.x, lane = tid & 31, wave = tid >> 5;
    const int zo = blockIdx.z / innerCount, zi = blockIdx.z % innerCount;
    A  += zo * aOut + zi * aIn;
    Bt += zo * bOut + zi * bIn;
    if constexpr (ADDIN) Add += zo * dOut + zi * dIn;
    const long long cBase = zo * cOut + zi * cIn;
    const int row0 = blockIdx.y * 64, col0 = blockIdx.x * 64;

    const int lr = tid >> 2;         // staging: row 0..63
    const int lc = (tid & 3) * 8;    // staging: k chunk
    const int smr  = (wave * 2) >> 2;       // subtile row 0..3
    const int snc  = (wave * 2) & 3;        // subtile col (even)
    const int arow = smr * 16 + (lane & 15);
    const int br0  = snc * 16 + (lane & 15);
    const int br1  = (snc + 1) * 16 + (lane & 15);
    const int koff = (lane >> 4) * 8;

    f32x8 acc0 = {}, acc1 = {};
    for (int k0 = 0; k0 < K; k0 += 32) {
        async_copy_b128(&As[lr * 32 + lc], &A [(long long)(row0 + lr) * ldA + k0 + lc]);
        async_copy_b128(&Bs[lr * 32 + lc], &Bt[(long long)(col0 + lr) * ldB + k0 + lc]);
        if (k0 + 32 < K) {  // prefetch next K chunk toward L2/WGP$
            __builtin_prefetch(&A [(long long)(row0 + lr) * ldA + k0 + 32 + lc], 0, 1);
            __builtin_prefetch(&Bt[(long long)(col0 + lr) * ldB + k0 + 32 + lc], 0, 1);
        }
        async_wait0();
        __syncthreads();
        ABFrag a, b0, b1;
        a.h[0]  = *(const bf16x8*)&As[arow * 32 + koff];
        a.h[1]  = *(const bf16x8*)&As[arow * 32 + koff + 16];
        b0.h[0] = *(const bf16x8*)&Bs[br0 * 32 + koff];
        b0.h[1] = *(const bf16x8*)&Bs[br0 * 32 + koff + 16];
        b1.h[0] = *(const bf16x8*)&Bs[br1 * 32 + koff];
        b1.h[1] = *(const bf16x8*)&Bs[br1 * 32 + koff + 16];
        acc0 = wmma_bf16(a, b0, acc0);
        acc1 = wmma_bf16(a, b1, acc1);
        __syncthreads();
    }
    const int mBase = row0 + smr * 16 + ((lane >> 4) << 3);
#pragma unroll
    for (int t = 0; t < 2; ++t) {
        const f32x8 acc = t ? acc1 : acc0;
        const int gc = col0 + (snc + t) * 16 + (lane & 15);
#pragma unroll
        for (int r = 0; r < 8; ++r) {
            const long long mr = mBase + r;
            float v = acc[r] * alpha;
            if constexpr (BIAS)  v += bias[gc];
            if constexpr (GELU)  v = 0.5f * v * (1.0f + erff(v * 0.70710678118654752f));
            if constexpr (ADDIN) v += Add[mr * ldAdd + gc];
            if constexpr (BF16OUT) ((bf16*)Cp)[cBase + mr * ldC + gc] = (bf16)v;
            else                   ((float*)Cp)[cBase + mr * ldC + gc] = v;
        }
    }
}

// ---------------------------------------------------------------------------
// Fused per-(b,i) kernel: reads rel_i = rel[b,i,:,:] (512x64 fp32) EXACTLY once.
// Dynamic LDS partition (~117 KB, well under CDNA5's 320 KB/WG):
//   qp   16x64  bf16 (2 KB)   : q rows padded to 16
//   sc   16x512 f32  (32 KB)  : scores (rows 6..15 write-only padding)
//   p    16x512 bf16 (16 KB)  : softmax probs, padded
//   relT 64x512 bf16 (64 KB)  : rel_i^T, filled during bias pass
// Pass 1: sc += qpad @ rel_i^T (WMMA); converted rel frags scattered to relT.
// Pass 2: softmax rows 0..5.
// Pass 3: O2 = p @ rel_i via WMMA with B-frags read contiguously from relT.
// 256 threads = 8 wave32.
// ---------------------------------------------------------------------------
__global__ __launch_bounds__(256)
void attn_rel_kernel(const float* __restrict__ rel,  // (B,N,N,64) fp32
                     const bf16*  __restrict__ qbf,  // (B,N,384) bf16, pre-scaled
                     const float* __restrict__ S,    // (B,H,N,N) fp32 (q@k^T)
                     bf16* __restrict__ P,           // (B,H,N,N) bf16 probs out
                     float* __restrict__ O2) {       // (B,N,384) fp32 out
    extern __shared__ __align__(16) char smem[];
    bf16*  qp   = (bf16*)smem;                       // 2048 B
    bf16*  p    = (bf16*)(smem + 2048);              // 16384 B
    bf16*  relT = (bf16*)(smem + 2048 + 16384);      // 65536 B
    float* sc   = (float*)(smem + 2048 + 16384 + 65536); // 32768 B

    const int tid = threadIdx.x, lane = tid & 31, wave = tid >> 5;
    const int bi = blockIdx.x;      // b*512 + i
    const int b = bi >> 9, i = bi & 511;

    for (int idx = tid; idx < 16 * 64; idx += 256) {
        const int h = idx >> 6, c = idx & 63;
        qp[idx] = (h < 6) ? qbf[(long long)bi * 384 + h * 64 + c] : (bf16)0.0f;
    }
    for (int idx = tid; idx < 16 * 512; idx += 256) p[idx] = (bf16)0.0f;
    // coalesced preload of scores: sc[h][j] = S[b,h,i,j]  (rows 0..5 only)
    for (int idx = tid; idx < (6 * 512) / 4; idx += 256) {
        const int h = idx >> 7;            // /128 (128 float4 per row)
        const int j4 = (idx & 127) * 4;
        *(float4*)&sc[h * 512 + j4] =
            *(const float4*)&S[((long long)(b * 6 + h) * 512 + i) * 512 + j4];
    }
    __syncthreads();

    const long long relBase = (long long)bi * 512 * 64;
    const int koff = (lane >> 4) * 8;
    const int arow = lane & 15;

    // --- 1. bias GEMM: 32 column tiles of 16; each wave does 4.
    //        Also scatters converted bf16 rel into relT (transposed). --------
    for (int t = 0; t < 4; ++t) {
        const int j0 = (wave * 4 + t) * 16;
        const int j  = j0 + (lane & 15);
        if (t < 3)  // prefetch next tile's rel rows
            __builtin_prefetch(rel + relBase + (long long)(j + 16) * 64, 0, 1);
        f32x8 acc = {};
        for (int k0 = 0; k0 < 64; k0 += 32) {
            ABFrag a, bb;
            a.h[0] = *(const bf16x8*)&qp[arow * 64 + k0 + koff];
            a.h[1] = *(const bf16x8*)&qp[arow * 64 + k0 + koff + 16];
            const float* rp = rel + relBase + (long long)j * 64 + k0 + koff;
#pragma unroll
            for (int e = 0; e < 8; ++e) bb.e[e]     = (bf16)rp[e];
#pragma unroll
            for (int e = 0; e < 8; ++e) bb.e[8 + e] = (bf16)rp[16 + e];
            // transpose-scatter into relT[c][j]
#pragma unroll
            for (int e = 0; e < 8; ++e) relT[(k0 + koff + e) * 512 + j]      = bb.e[e];
#pragma unroll
            for (int e = 0; e < 8; ++e) relT[(k0 + koff + 16 + e) * 512 + j] = bb.e[8 + e];
            acc = wmma_bf16(a, bb, acc);
        }
        // unconditional accumulate (rows 6..15 of sc are padding)
#pragma unroll
        for (int r = 0; r < 8; ++r) {
            const int m = r + ((lane >> 4) << 3);
            sc[m * 512 + j] += acc[r];
        }
    }
    __syncthreads();

    // --- 2. softmax: wave w handles head h=w (waves 6,7 idle) --------------
    if (wave < 6) {
        const int h = wave;
        float vals[16], mx = -3.4e38f;
#pragma unroll
        for (int t = 0; t < 16; ++t) { vals[t] = sc[h * 512 + lane + t * 32]; mx = fmaxf(mx, vals[t]); }
        for (int o = 16; o > 0; o >>= 1) mx = fmaxf(mx, __shfl_xor(mx, o, 32));
        float sum = 0.f;
#pragma unroll
        for (int t = 0; t < 16; ++t) { vals[t] = expf(vals[t] - mx); sum += vals[t]; }
        for (int o = 16; o > 0; o >>= 1) sum += __shfl_xor(sum, o, 32);
        const float inv = 1.0f / sum;
        bf16* Prow = P + ((long long)(b * 6 + h) * 512 + i) * 512;
#pragma unroll
        for (int t = 0; t < 16; ++t) {
            const bf16 pb = (bf16)(vals[t] * inv);
            p[h * 512 + lane + t * 32] = pb;
            Prow[lane + t * 32] = pb;
        }
    }
    __syncthreads();

    // --- 3. O2 = p(16x512) @ rel_i(512x64); waves 0..3 each 16 columns.
    //        B-fragments are contiguous rows of relT (two ds_load_b128). ----
    if (wave < 4) {
        const int c0 = wave * 16;
        const int n  = c0 + (lane & 15);
        f32x8 acc = {};
        for (int k0 = 0; k0 < 512; k0 += 32) {
            ABFrag a, bb;
            a.h[0]  = *(const bf16x8*)&p[arow * 512 + k0 + koff];
            a.h[1]  = *(const bf16x8*)&p[arow * 512 + k0 + koff + 16];
            bb.h[0] = *(const bf16x8*)&relT[n * 512 + k0 + koff];
            bb.h[1] = *(const bf16x8*)&relT[n * 512 + k0 + koff + 16];
            acc = wmma_bf16(a, bb, acc);
        }
#pragma unroll
        for (int r = 0; r < 8; ++r) {
            const int m = r + ((lane >> 4) << 3);
            if (m < 6) O2[(long long)bi * 384 + m * 64 + n] = acc[r];
        }
    }
}

// ---------------------------------------------------------------------------
extern "C" void kernel_launch(void* const* d_in, const int* in_sizes, int n_in,
                              void* d_out, int out_size, void* d_ws, size_t ws_size,
                              hipStream_t stream) {
    (void)in_sizes; (void)n_in; (void)out_size; (void)ws_size;
    const float* x    = (const float*)d_in[0];
    const float* rel  = (const float*)d_in[1];
    const float* ln1w = (const float*)d_in[2];
    const float* ln1b = (const float*)d_in[3];
    const float* ln2w = (const float*)d_in[4];
    const float* ln2b = (const float*)d_in[5];
    const float* Wq   = (const float*)d_in[6];
    const float* Wk   = (const float*)d_in[7];
    const float* Wv   = (const float*)d_in[8];
    const float* Wo   = (const float*)d_in[9];
    const float* bo   = (const float*)d_in[10];
    const float* W1   = (const float*)d_in[11];
    const float* b1   = (const float*)d_in[12];
    const float* W2   = (const float*)d_in[13];
    const float* b2   = (const float*)d_in[14];

    const int BN = 4 * 512, D = 384, FF = 1536, N = 512, H = 6;
    char* ws = (char*)d_ws;
    size_t off = 0;
    auto alloc = [&](size_t bytes) -> char* {
        char* p = ws + off; off += (bytes + 255) & ~(size_t)255; return p;
    };
    bf16*  xnb  = (bf16*)alloc((size_t)BN * D * 2);
    bf16*  qb   = (bf16*)alloc((size_t)BN * D * 2);
    bf16*  kb   = (bf16*)alloc((size_t)BN * D * 2);
    bf16*  vb   = (bf16*)alloc((size_t)BN * D * 2);
    bf16*  vT   = (bf16*)alloc((size_t)BN * D * 2);
    bf16*  Ob   = (bf16*)alloc((size_t)BN * D * 2);
    bf16*  xn2b = (bf16*)alloc((size_t)BN * D * 2);
    bf16*  hb   = (bf16*)alloc((size_t)BN * FF * 2);
    bf16*  Wqb  = (bf16*)alloc((size_t)D * D * 2);
    bf16*  Wkb  = (bf16*)alloc((size_t)D * D * 2);
    bf16*  Wvb  = (bf16*)alloc((size_t)D * D * 2);
    bf16*  Wob  = (bf16*)alloc((size_t)D * D * 2);
    bf16*  W1b  = (bf16*)alloc((size_t)FF * D * 2);
    bf16*  W2b  = (bf16*)alloc((size_t)D * FF * 2);
    float* Sf   = (float*)alloc((size_t)4 * H * N * N * 4);
    bf16*  Pb   = (bf16*)alloc((size_t)4 * H * N * N * 2);
    float* O2f  = (float*)alloc((size_t)BN * D * 4);
    float* x1   = (float*)alloc((size_t)BN * D * 4);

    // weights -> bf16
    cvt_bf16_kernel<<<256, 256, 0, stream>>>(Wq, Wqb, D * D);
    cvt_bf16_kernel<<<256, 256, 0, stream>>>(Wk, Wkb, D * D);
    cvt_bf16_kernel<<<256, 256, 0, stream>>>(Wv, Wvb, D * D);
    cvt_bf16_kernel<<<256, 256, 0, stream>>>(Wo, Wob, D * D);
    cvt_bf16_kernel<<<512, 256, 0, stream>>>(W1, W1b, FF * D);
    cvt_bf16_kernel<<<512, 256, 0, stream>>>(W2, W2b, D * FF);

    // LN1
    ln_kernel<<<BN, 128, 0, stream>>>(x, ln1w, ln1b, xnb);

    // q (pre-scaled by HD^-0.5), k, v projections
    gemm_bf16_kernel<false,false,true,false><<<dim3(D/64, BN/64, 1), 256, 0, stream>>>(
        xnb, D, 0, 0, Wqb, D, 0, 0, qb, D, 0, 0, nullptr, nullptr, 0, 0, 0, 1, D, 0.125f);
    gemm_bf16_kernel<false,false,true,false><<<dim3(D/64, BN/64, 1), 256, 0, stream>>>(
        xnb, D, 0, 0, Wkb, D, 0, 0, kb, D, 0, 0, nullptr, nullptr, 0, 0, 0, 1, D, 1.0f);
    gemm_bf16_kernel<false,false,true,false><<<dim3(D/64, BN/64, 1), 256, 0, stream>>>(
        xnb, D, 0, 0, Wvb, D, 0, 0, vb, D, 0, 0, nullptr, nullptr, 0, 0, 0, 1, D, 1.0f);

    // S[b,h] = q_slice @ k_slice^T  (24 batches, 512x512x64)
    gemm_bf16_kernel<false,false,false,false><<<dim3(N/64, N/64, 4 * H), 256, 0, stream>>>(
        qb, D, (long long)N * D, 64,
        kb, D, (long long)N * D, 64,
        Sf, N, (long long)H * N * N, (long long)N * N,
        nullptr, nullptr, 0, 0, 0, H, 64, 1.0f);

    // vT (B,H,64,N)
    transpose_v_kernel<<<3072, 256, 0, stream>>>(vb, vT, 4 * H * 64 * N);

    // fused: rel bias + softmax + probs@rel  (single pass over 256MB rel)
    const int relSmem = 2048 + 16384 + 65536 + 32768;   // ~114 KB dynamic LDS
    attn_rel_kernel<<<BN, 256, relSmem, stream>>>(rel, qb, Sf, Pb, O2f);

    // O = P @ v + O2  ->  Ob bf16 (B,N,384)
    gemm_bf16_kernel<false,false,true,true><<<dim3(1, N/64, 4 * H), 256, 0, stream>>>(
        Pb, N, (long long)H * N * N, (long long)N * N,
        vT, N, (long long)H * 64 * N, (long long)64 * N,
        Ob, D, (long long)N * D, 64,
        nullptr, O2f, D, (long long)N * D, 64, H, N, 1.0f);

    // x1 = x + O @ Wo^T + bo
    gemm_bf16_kernel<true,false,false,true><<<dim3(D/64, BN/64, 1), 256, 0, stream>>>(
        Ob, D, 0, 0, Wob, D, 0, 0, x1, D, 0, 0, bo, x, D, 0, 0, 1, D, 1.0f);

    // LN2
    ln_kernel<<<BN, 128, 0, stream>>>(x1, ln2w, ln2b, xn2b);

    // h = gelu(xn2 @ W1^T + b1)
    gemm_bf16_kernel<true,true,true,false><<<dim3(FF/64, BN/64, 1), 256, 0, stream>>>(
        xn2b, D, 0, 0, W1b, D, 0, 0, hb, FF, 0, 0, b1, nullptr, 0, 0, 0, 1, D, 1.0f);

    // out = x1 + h @ W2^T + b2
    gemm_bf16_kernel<true,false,false,true><<<dim3(D/64, BN/64, 1), 256, 0, stream>>>(
        hb, FF, 0, 0, W2b, FF, 0, 0, (float*)d_out, D, 0, 0, b2, x1, D, 0, 0, 1, FF, 1.0f);
}
